// GRU_6339371729228
// MI455X (gfx1250) — matrix operand order repaired and checked by reading, not compiled, not run on hardware
//
#include <hip/hip_runtime.h>

typedef unsigned short u16;
typedef unsigned long long u64;
typedef __attribute__((ext_vector_type(16))) __bf16 v16bf;
typedef __attribute__((ext_vector_type(8)))  float  v8f;

#define GRU_H 1024
#define GRU_N 16384
#define TM 128       // rows per workgroup
#define TN 32        // cols per workgroup
#define TK 32        // K step = bf16 WMMA K
#define TKP 40       // padded LDS pitch (80B: 16B-aligned b128, 20-bank stride)
#define NKSTEPS (GRU_H / TK)

union ABFrag { uint4 q[2]; v16bf v; };

#define WMMA_BF16(A, B, C) \
  __builtin_amdgcn_wmma_f32_16x16x32_bf16(false, (A), false, (B), (short)0, (C), false, false)

#if defined(__has_builtin)
#if __has_builtin(__builtin_amdgcn_s_wait_asynccnt)
#define WAIT_ASYNC(n) __builtin_amdgcn_s_wait_asynccnt(n)
#endif
#endif
#ifndef WAIT_ASYNC
#define WAIT_ASYNC(n) asm volatile("s_wait_asynccnt %0" :: "i"(n) : "memory")
#endif

// CDNA5 async DMA: global -> LDS, 16B per lane, tracked by ASYNCcnt.
// SADDR form: global_load_async_to_lds_b128 vDstLds, vOffset, s[base:base+1]
__device__ __forceinline__ void async_b128(const u16* lds_dst, const u16* gbase,
                                           unsigned byte_off) {
  asm volatile("global_load_async_to_lds_b128 %0, %1, %2"
               :: "v"((unsigned)(u64)lds_dst), "v"(byte_off), "s"((u64)gbase)
               : "memory");
}

__device__ __forceinline__ u16 f2bf(float f) {
  unsigned u = __float_as_uint(f);
  unsigned r = u + 0x7FFFu + ((u >> 16) & 1u);
  return (u16)(r >> 16);
}
__device__ __forceinline__ float bf2f(u16 b) {
  return __uint_as_float(((unsigned)b) << 16);
}
__device__ __forceinline__ float sigmoidf(float x) {
  return 1.0f / (1.0f + __expf(-x));
}

// ---------------- fp32 -> bf16 pack (8 elems / thread) ----------------
__global__ __launch_bounds__(256) void cvt_f32_bf16(const float* __restrict__ src,
                                                    u16* __restrict__ dst, int n8) {
  int i = blockIdx.x * blockDim.x + threadIdx.x;
  if (i >= n8) return;
  const float4* s = (const float4*)src;
  float4 a = s[i * 2 + 0];
  float4 b = s[i * 2 + 1];
  uint4 o;
  o.x = (unsigned)f2bf(a.x) | ((unsigned)f2bf(a.y) << 16);
  o.y = (unsigned)f2bf(a.z) | ((unsigned)f2bf(a.w) << 16);
  o.z = (unsigned)f2bf(b.x) | ((unsigned)f2bf(b.y) << 16);
  o.w = (unsigned)f2bf(b.z) | ((unsigned)f2bf(b.w) << 16);
  ((uint4*)dst)[i] = o;
}

// ---------------- fused z/r gates: 4 GEMMs + sigmoid + r*hidden ----------------
__global__ __launch_bounds__(256) void gru_zr(
    const u16* __restrict__ uBf, const u16* __restrict__ hBf,
    const u16* __restrict__ Wz, const u16* __restrict__ Uz,
    const u16* __restrict__ Wr, const u16* __restrict__ Ur,
    const float* __restrict__ bz, const float* __restrict__ br,
    const float* __restrict__ hidF,
    u16* __restrict__ zBf, u16* __restrict__ rhBf)
{
  __shared__ u16 lA[2][2][TM * TKP];   // [buf][U/Hd]
  __shared__ u16 lB[2][4][TN * TKP];   // [buf][Wz/Uz/Wr/Ur]

  const int tid  = threadIdx.x;
  const int wv   = tid >> 5;           // wave32 id: owns rows wv*16..wv*16+15
  const int lane = tid & 31;
  const int hf   = lane >> 4;
  const int nl   = lane & 15;
  const int row0 = blockIdx.y * TM;
  const int col0 = blockIdx.x * TN;

  // per-thread staging coords (uniform 6 async ops / thread / stage)
  const int ar = tid >> 2;             // A row chunk base (0..63)
  const int ac = (tid & 3) << 3;       // A k-col
  const int bt0 = (2 * tid) >> 7;      // B tile for s=0
  const int bi0 = (2 * tid) & 127;
  const int bt1 = (2 * tid + 1) >> 7;
  const int bi1 = (2 * tid + 1) & 127;

  auto stage = [&](int buf, int kk) {
#pragma unroll
    for (int it = 0; it < 2; ++it) {
      int r = ar + it * 64;
      unsigned go = (unsigned)(((row0 + r) * GRU_H + kk + ac) * 2);
      async_b128(&lA[buf][0][r * TKP + ac], uBf, go);
      async_b128(&lA[buf][1][r * TKP + ac], hBf, go);
    }
    const u16* wptr[4] = {Wz, Uz, Wr, Ur};
    {
      int r = bi0 >> 2, c = (bi0 & 3) << 3;
      unsigned go = (unsigned)(((col0 + r) * GRU_H + kk + c) * 2);
      async_b128(&lB[buf][bt0][r * TKP + c], wptr[bt0], go);
    }
    {
      int r = bi1 >> 2, c = (bi1 & 3) << 3;
      unsigned go = (unsigned)(((col0 + r) * GRU_H + kk + c) * 2);
      async_b128(&lB[buf][bt1][r * TKP + c], wptr[bt1], go);
    }
  };

  const v8f vzero = {0.f, 0.f, 0.f, 0.f, 0.f, 0.f, 0.f, 0.f};
  v8f accz[2], accr[2];
#pragma unroll
  for (int n = 0; n < 2; ++n) { accz[n] = vzero; accr[n] = vzero; }

  stage(0, 0);
  int p = 0;
  for (int i = 0; i < NKSTEPS; ++i) {
    if (i + 1 < NKSTEPS) { stage(p ^ 1, (i + 1) * TK); WAIT_ASYNC(6); }
    else                 { WAIT_ASYNC(0); }
    __syncthreads();

    // A frags (16x32): half 0 -> K 0-7 & 16-23, half 1 -> K 8-15 & 24-31
    const int abase = (wv * 16 + nl) * TKP + hf * 8;
    ABFrag aU, aH;
    aU.q[0] = *(const uint4*)&lA[p][0][abase];
    aU.q[1] = *(const uint4*)&lA[p][0][abase + 16];
    aH.q[0] = *(const uint4*)&lA[p][1][abase];
    aH.q[1] = *(const uint4*)&lA[p][1][abase + 16];

#pragma unroll
    for (int n = 0; n < 2; ++n) {
      const int bbase = (n * 16 + nl) * TKP + hf * 8;
      ABFrag b0, b1, b2, b3;
      b0.q[0] = *(const uint4*)&lB[p][0][bbase]; b0.q[1] = *(const uint4*)&lB[p][0][bbase + 16];
      b1.q[0] = *(const uint4*)&lB[p][1][bbase]; b1.q[1] = *(const uint4*)&lB[p][1][bbase + 16];
      b2.q[0] = *(const uint4*)&lB[p][2][bbase]; b2.q[1] = *(const uint4*)&lB[p][2][bbase + 16];
      b3.q[0] = *(const uint4*)&lB[p][3][bbase]; b3.q[1] = *(const uint4*)&lB[p][3][bbase + 16];
      accz[n] = WMMA_BF16(aU.v, b0.v, accz[n]);
      accz[n] = WMMA_BF16(aH.v, b1.v, accz[n]);
      accr[n] = WMMA_BF16(aU.v, b2.v, accr[n]);
      accr[n] = WMMA_BF16(aH.v, b3.v, accr[n]);
    }
    __syncthreads();
    p ^= 1;
  }

  const int mbase = row0 + wv * 16 + hf * 8;
#pragma unroll
  for (int n = 0; n < 2; ++n) {
    const int gc = col0 + n * 16 + nl;
    const float bzv = bz[gc];
    const float brv = br[gc];
#pragma unroll
    for (int j = 0; j < 8; ++j) {
      const size_t gi = (size_t)(mbase + j) * GRU_H + gc;
      float zv = sigmoidf(accz[n][j] + bzv);
      float rv = sigmoidf(accr[n][j] + brv);
      zBf[gi]  = f2bf(zv);
      rhBf[gi] = f2bf(rv * hidF[gi]);
    }
  }
}

// ---------------- h gate: 2 GEMMs + tanh + final blend ----------------
__global__ __launch_bounds__(256) void gru_h(
    const u16* __restrict__ uBf, const u16* __restrict__ rhBf,
    const u16* __restrict__ Wh, const u16* __restrict__ Uh,
    const float* __restrict__ bh, const float* __restrict__ hidF,
    const u16* __restrict__ zBf, float* __restrict__ out)
{
  __shared__ u16 lA[2][2][TM * TKP];
  __shared__ u16 lB[2][2][TN * TKP];

  const int tid  = threadIdx.x;
  const int wv   = tid >> 5;
  const int lane = tid & 31;
  const int hf   = lane >> 4;
  const int nl   = lane & 15;
  const int row0 = blockIdx.y * TM;
  const int col0 = blockIdx.x * TN;

  const int ar = tid >> 2;
  const int ac = (tid & 3) << 3;
  const int bt = tid >> 7;             // B tile (0..1)
  const int bi = tid & 127;

  auto stage = [&](int buf, int kk) {
#pragma unroll
    for (int it = 0; it < 2; ++it) {
      int r = ar + it * 64;
      unsigned go = (unsigned)(((row0 + r) * GRU_H + kk + ac) * 2);
      async_b128(&lA[buf][0][r * TKP + ac], uBf, go);
      async_b128(&lA[buf][1][r * TKP + ac], rhBf, go);
    }
    const u16* wptr[2] = {Wh, Uh};
    {
      int r = bi >> 2, c = (bi & 3) << 3;
      unsigned go = (unsigned)(((col0 + r) * GRU_H + kk + c) * 2);
      async_b128(&lB[buf][bt][r * TKP + c], wptr[bt], go);
    }
  };

  const v8f vzero = {0.f, 0.f, 0.f, 0.f, 0.f, 0.f, 0.f, 0.f};
  v8f acc[2];
#pragma unroll
  for (int n = 0; n < 2; ++n) acc[n] = vzero;

  stage(0, 0);
  int p = 0;
  for (int i = 0; i < NKSTEPS; ++i) {
    if (i + 1 < NKSTEPS) { stage(p ^ 1, (i + 1) * TK); WAIT_ASYNC(5); }
    else                 { WAIT_ASYNC(0); }
    __syncthreads();

    const int abase = (wv * 16 + nl) * TKP + hf * 8;
    ABFrag aU, aR;
    aU.q[0] = *(const uint4*)&lA[p][0][abase];
    aU.q[1] = *(const uint4*)&lA[p][0][abase + 16];
    aR.q[0] = *(const uint4*)&lA[p][1][abase];
    aR.q[1] = *(const uint4*)&lA[p][1][abase + 16];

#pragma unroll
    for (int n = 0; n < 2; ++n) {
      const int bbase = (n * 16 + nl) * TKP + hf * 8;
      ABFrag b0, b1;
      b0.q[0] = *(const uint4*)&lB[p][0][bbase]; b0.q[1] = *(const uint4*)&lB[p][0][bbase + 16];
      b1.q[0] = *(const uint4*)&lB[p][1][bbase]; b1.q[1] = *(const uint4*)&lB[p][1][bbase + 16];
      acc[n] = WMMA_BF16(aU.v, b0.v, acc[n]);
      acc[n] = WMMA_BF16(aR.v, b1.v, acc[n]);
    }
    __syncthreads();
    p ^= 1;
  }

  const int mbase = row0 + wv * 16 + hf * 8;
#pragma unroll
  for (int n = 0; n < 2; ++n) {
    const int gc = col0 + n * 16 + nl;
    const float bhv = bh[gc];
#pragma unroll
    for (int j = 0; j < 8; ++j) {
      const size_t gi = (size_t)(mbase + j) * GRU_H + gc;
      float hv  = tanhf(acc[n][j] + bhv);
      float zv  = bf2f(zBf[gi]);
      float hid = hidF[gi];
      out[gi] = (1.0f - zv) * hid + zv * hv;
    }
  }
}

extern "C" void kernel_launch(void* const* d_in, const int* in_sizes, int n_in,
                              void* d_out, int out_size, void* d_ws, size_t ws_size,
                              hipStream_t stream) {
  const float* update = (const float*)d_in[0];
  const float* hidden = (const float*)d_in[1];
  const float* wz = (const float*)d_in[2];
  const float* uz = (const float*)d_in[3];
  const float* bz = (const float*)d_in[4];
  const float* wr = (const float*)d_in[5];
  const float* ur = (const float*)d_in[6];
  const float* br = (const float*)d_in[7];
  const float* wh = (const float*)d_in[8];
  const float* uh = (const float*)d_in[9];
  const float* bh = (const float*)d_in[10];
  float* out = (float*)d_out;

  const size_t NH = (size_t)GRU_N * GRU_H;
  const size_t HH = (size_t)GRU_H * GRU_H;

  u16* uBf  = (u16*)d_ws;
  u16* hBf  = uBf  + NH;
  u16* wzB  = hBf  + NH;
  u16* uzB  = wzB  + HH;
  u16* wrB  = uzB  + HH;
  u16* urB  = wrB  + HH;
  u16* whB  = urB  + HH;
  u16* uhB  = whB  + HH;
  u16* rhBf = uhB  + HH;
  u16* zBf  = rhBf + NH;

  dim3 blk(256);
  cvt_f32_bf16<<<(int)(NH / 8 / 256), blk, 0, stream>>>(update, uBf, (int)(NH / 8));
  cvt_f32_bf16<<<(int)(NH / 8 / 256), blk, 0, stream>>>(hidden, hBf, (int)(NH / 8));
  cvt_f32_bf16<<<(int)(HH / 8 / 256), blk, 0, stream>>>(wz, wzB, (int)(HH / 8));
  cvt_f32_bf16<<<(int)(HH / 8 / 256), blk, 0, stream>>>(uz, uzB, (int)(HH / 8));
  cvt_f32_bf16<<<(int)(HH / 8 / 256), blk, 0, stream>>>(wr, wrB, (int)(HH / 8));
  cvt_f32_bf16<<<(int)(HH / 8 / 256), blk, 0, stream>>>(ur, urB, (int)(HH / 8));
  cvt_f32_bf16<<<(int)(HH / 8 / 256), blk, 0, stream>>>(wh, whB, (int)(HH / 8));
  cvt_f32_bf16<<<(int)(HH / 8 / 256), blk, 0, stream>>>(uh, uhB, (int)(HH / 8));

  dim3 grd(GRU_H / TN, GRU_N / TM);   // 32 x 128 workgroups
  gru_zr<<<grd, blk, 0, stream>>>(uBf, hBf, wzB, uzB, wrB, urB, bz, br, hidden,
                                  zBf, rhBf);
  gru_h<<<grd, blk, 0, stream>>>(uBf, rhBf, whB, uhB, bh, hidden, zBf, out);
}
